// MultiHeadAttention_83322365542430
// MI455X (gfx1250) — compile-verified, matrix operand order checked by reading, not run on hardware
//
#include <hip/hip_runtime.h>
#include <hip/hip_bf16.h>

// ---------------------------------------------------------------------------
// XCA (cross-covariance) attention on MI455X (gfx1250), bf16 WMMA pipeline.
//   b=8, n=4096, c=1024, H=16, d=64.  ~283 GFLOP, L2-resident (192MB L2).
//   Compute-bound: bf16 WMMA (v_wmma_f32_16x16x32_bf16) everywhere.
// v2: double-buffered LDS tiles (1 barrier/K-step), 2 N-tiles per wave
//   (8 wmma per K-step per wave), packed 16B Q/K stores, async global->LDS
//   staging (ASYNCcnt) in the output GEMM.
// ---------------------------------------------------------------------------

typedef __attribute__((ext_vector_type(16))) __bf16 v16bf;
typedef __attribute__((ext_vector_type(8)))  __bf16 v8bf;
typedef __attribute__((ext_vector_type(8)))  float  v8f;

#if __has_builtin(__builtin_amdgcn_global_load_async_to_lds_b128)
#define HAVE_ASYNC_LDS 1
typedef int v4i_vs __attribute__((vector_size(16)));  // matches builtin param
#endif

// A/B fragment for v_wmma_f32_16x16x32_bf16 from a row-major [rows x ld] bf16
// matrix. Lane L (0..15): row = row0+L, K = k0+0..7 and k0+16..23.
// Lane L (16..31): row = row0+L-16, K = k0+8..15 and k0+24..31.
static __device__ inline v16bf load_frag_bf16(const __bf16* __restrict__ p,
                                              int ld, int row0, int k0) {
  const int lane = threadIdx.x & 31;
  const __bf16* base = p + (size_t)(row0 + (lane & 15)) * (size_t)ld
                         + (size_t)(k0 + ((lane >> 4) << 3));
  union { v16bf v; v8bf h[2]; } u;
  u.h[0] = *(const v8bf*)(base);        // K = k..k+7
  u.h[1] = *(const v8bf*)(base + 16);   // K = k+16..k+23
  return u.v;
}

static __device__ inline v8f wmma_bf16(v16bf a, v16bf b, v8f c) {
  return __builtin_amdgcn_wmma_f32_16x16x32_bf16(
      /*neg_a=*/false, a, /*neg_b=*/false, b,
      /*c_mod=*/(short)0, c, /*reuse_a=*/false, /*reuse_b=*/false);
}

// ---------------------------------------------------------------------------
// K0: weights fp32 -> bf16.  grid (1024, 4) x 256, 4 elems/thread.
__global__ __launch_bounds__(256) void cvt_w_kernel(
    const float* __restrict__ Wq, const float* __restrict__ Wk,
    const float* __restrict__ Wv, const float* __restrict__ Wo,
    __bf16* __restrict__ Wqkv, __bf16* __restrict__ Wob) {
  const int which = blockIdx.y;
  const float* src = which == 0 ? Wq : which == 1 ? Wk : which == 2 ? Wv : Wo;
  __bf16* dst = which < 3 ? (Wqkv + (size_t)which * 1048576) : Wob;
  size_t i = ((size_t)blockIdx.x * 256 + threadIdx.x) * 4;
#pragma unroll
  for (int j = 0; j < 4; ++j) dst[i + j] = (__bf16)src[i + j];
}

// ---------------------------------------------------------------------------
// K1: fused QKV projection.  C[32768 x 3072] = x[32768 x 1024] @ Wqkv^T.
// WG = 256 thr (8 waves) computes a 64x256 tile; wave w owns 2 N-col blocks.
// x tile (64x32) staged fp32->bf16 in double-buffered LDS (1 barrier/step).
__global__ __launch_bounds__(256) void proj_kernel(
    const float* __restrict__ x, const __bf16* __restrict__ Wqkv,
    __bf16* __restrict__ Qt, __bf16* __restrict__ Kt, __bf16* __restrict__ Vt) {
  __shared__ __bf16 As[2][64][40];  // 32 cols + pad (80B rows, 16B aligned)
  const int lane = threadIdx.x & 31;
  const int wave = threadIdx.x >> 5;
  const int row0 = blockIdx.x * 64;               // token tile
  const int col0 = blockIdx.y * 256 + wave * 32;  // out-channel (0..3071)

  const int r = threadIdx.x >> 2;        // 0..63
  const int c = (threadIdx.x & 3) * 8;   // 0,8,16,24
  const float* xrow = x + (size_t)(row0 + r) * 1024 + c;

  v8f acc[4][2] = {};

  {  // preload K-tile 0 into buffer 0
    v8bf t;
#pragma unroll
    for (int i = 0; i < 8; ++i) t[i] = (__bf16)xrow[i];
    *(v8bf*)&As[0][r][c] = t;
  }
  __syncthreads();

  for (int k0 = 0; k0 < 1024; k0 += 32) {
    const int buf = (k0 >> 5) & 1;
    const bool more = (k0 + 32) < 1024;
    float nf[8];
    if (more) {  // issue next-tile global loads early; overlap with wmma
      const float* src = xrow + k0 + 32;
      __builtin_prefetch(src + 32, 0, 3);  // K-step after next
#pragma unroll
      for (int i = 0; i < 8; ++i) nf[i] = src[i];
    }
    v16bf b0 = load_frag_bf16(Wqkv, 1024, col0, k0);
    v16bf b1 = load_frag_bf16(Wqkv, 1024, col0 + 16, k0);
#pragma unroll
    for (int mi = 0; mi < 4; ++mi) {
      v16bf a = load_frag_bf16(&As[buf][0][0], 40, mi * 16, 0);
      acc[mi][0] = wmma_bf16(a, b0, acc[mi][0]);
      acc[mi][1] = wmma_bf16(a, b1, acc[mi][1]);
    }
    if (more) {
      v8bf t;
#pragma unroll
      for (int i = 0; i < 8; ++i) t[i] = (__bf16)nf[i];
      *(v8bf*)&As[buf ^ 1][r][c] = t;
    }
    __syncthreads();
  }

  // Store into transposed head layouts.
  const int rr0 = (lane >> 4) << 3;
#pragma unroll
  for (int nj = 0; nj < 2; ++nj) {
    const int cj = col0 + nj * 16;
    const int which = cj >> 10;             // 0=Q 1=K 2=V
    const int cc = cj & 1023;
    const int h = cc >> 6;
    const int dd = (cc & 63) + (lane & 15); // channel within head
#pragma unroll
    for (int mi = 0; mi < 4; ++mi) {
      const int m = row0 + mi * 16 + rr0;   // first of 8 consecutive tokens
      const int b = m >> 12, nn = m & 4095;
      const size_t bh = (size_t)b * 16 + h;
      if (which == 2) {  // V: [b,h,n,d] -> transpose scatter (2B stores)
#pragma unroll
        for (int i = 0; i < 8; ++i)
          Vt[(bh * 4096 + (nn + i)) * 64 + dd] = (__bf16)acc[mi][nj][i];
      } else {           // Q/K: [b,h,d,n] -> 8 consecutive tokens = 16B store
        v8bf pk;
#pragma unroll
        for (int i = 0; i < 8; ++i) pk[i] = (__bf16)acc[mi][nj][i];
        __bf16* dstp = (which == 0 ? Qt : Kt) + (bh * 64 + dd) * 4096 + nn;
        *(v8bf*)dstp = pk;
      }
    }
  }
}

// ---------------------------------------------------------------------------
// K2: inverse L2 norm of each [b,h,d] row over n=4096.  1 wave per row.
__global__ __launch_bounds__(256) void norm_kernel(
    const __bf16* __restrict__ Qt, const __bf16* __restrict__ Kt,
    float* __restrict__ invq, float* __restrict__ invk) {
  const int lane = threadIdx.x & 31;
  const int wave = threadIdx.x >> 5;
  const int row = blockIdx.x * 8 + wave;  // 0..16383 (first 8192 = Q)
  const __bf16* rp = (row < 8192 ? Qt : Kt) + (size_t)(row & 8191) * 4096;
  float s = 0.f;
  for (int base = 0; base < 4096; base += 256) {
    v8bf v = *(const v8bf*)(rp + base + lane * 8);
#pragma unroll
    for (int i = 0; i < 8; ++i) { float f = (float)v[i]; s += f * f; }
  }
#pragma unroll
  for (int off = 16; off >= 1; off >>= 1) s += __shfl_xor(s, off, 32);
  if (lane == 0) {
    float inv = 1.0f / fmaxf(sqrtf(s), 1e-12f);
    (row < 8192 ? invq : invk)[row & 8191] = inv;
  }
}

// ---------------------------------------------------------------------------
// K3: per-(b,h) attention block.  WG = 256 thr (8 waves), one per (b,h).
__global__ __launch_bounds__(256) void attn_kernel(
    const __bf16* __restrict__ Qt, const __bf16* __restrict__ Kt,
    const __bf16* __restrict__ Vt, const float* __restrict__ invq,
    const float* __restrict__ invk, const float* __restrict__ scale,
    __bf16* __restrict__ Obf) {
  __shared__ float  Sm[64][65];   // logits / probs (f32)
  __shared__ __bf16 Pm[64][72];   // probs bf16 (144B rows, 16B aligned)
  const int lane = threadIdx.x & 31;
  const int wave = threadIdx.x >> 5;
  const int bh = blockIdx.x;                 // 0..127
  const __bf16* Kbase = Kt + (size_t)bh * 64 * 4096;
  const __bf16* Qbase = Qt + (size_t)bh * 64 * 4096;

  // ---- Phase 1: S = K(64x4096) @ Q^T -> 4x4 grid of 16x16 tiles, 2/wave.
  const int t0 = wave * 2;
  const int mi = t0 >> 2;
  const int ni0 = t0 & 3, ni1 = ni0 + 1;
  v8f a0 = {}, a1 = {};
  for (int k0 = 0; k0 < 4096; k0 += 32) {
    v16bf af = load_frag_bf16(Kbase, 4096, mi * 16, k0);
    v16bf b0 = load_frag_bf16(Qbase, 4096, ni0 * 16, k0);
    v16bf b1 = load_frag_bf16(Qbase, 4096, ni1 * 16, k0);
    a0 = wmma_bf16(af, b0, a0);
    a1 = wmma_bf16(af, b1, a1);
  }
  {
    const float sc = scale[bh & 15];
    const float* iq = invq + bh * 64;
    const float* ik = invk + bh * 64;
    const int colA = ni0 * 16 + (lane & 15);
    const int colB = ni1 * 16 + (lane & 15);
    const int rbase = mi * 16 + ((lane >> 4) << 3);
#pragma unroll
    for (int i = 0; i < 8; ++i) {
      const int rr = rbase + i;
      Sm[rr][colA] = a0[i] * ik[rr] * iq[colA] * sc;
      Sm[rr][colB] = a1[i] * ik[rr] * iq[colB] * sc;
    }
  }
  __syncthreads();

  // ---- Phase 2: row softmax (64 rows), write bf16 probs.
  if (threadIdx.x < 64) {
    const int rr = threadIdx.x;
    float mx = -3.4e38f;
    for (int j = 0; j < 64; ++j) mx = fmaxf(mx, Sm[rr][j]);
    float sum = 0.f;
    for (int j = 0; j < 64; ++j) { float e = __expf(Sm[rr][j] - mx); Sm[rr][j] = e; sum += e; }
    const float inv = 1.0f / sum;
    for (int j = 0; j < 64; ++j) Pm[rr][j] = (__bf16)(Sm[rr][j] * inv);
  }
  __syncthreads();

  // ---- Phase 3: O(64x4096) = P(64x64) @ V(64x4096); Vt rows are [n][e].
  v16bf Af[4][2];
#pragma unroll
  for (int m2 = 0; m2 < 4; ++m2)
#pragma unroll
    for (int kk = 0; kk < 2; ++kk)
      Af[m2][kk] = load_frag_bf16(&Pm[0][0], 72, m2 * 16, kk * 32);

  const __bf16* Vbase = Vt + (size_t)bh * 4096 * 64;
  __bf16* Obase = Obf + (size_t)bh * 64 * 4096;
  for (int nt = wave; nt < 256; nt += 8) {
    const int n0 = nt * 16;
    v16bf B0 = load_frag_bf16(Vbase, 64, n0, 0);
    v16bf B1 = load_frag_bf16(Vbase, 64, n0, 32);
#pragma unroll
    for (int m2 = 0; m2 < 4; ++m2) {
      v8f acc = {};
      acc = wmma_bf16(Af[m2][0], B0, acc);
      acc = wmma_bf16(Af[m2][1], B1, acc);
      const int col = n0 + (lane & 15);
      const int rb = m2 * 16 + ((lane >> 4) << 3);
#pragma unroll
      for (int i = 0; i < 8; ++i)
        Obase[(size_t)(rb + i) * 4096 + col] = (__bf16)acc[i];
    }
  }
}

// ---------------------------------------------------------------------------
// K4: Y[32768 x 1024] = O_view @ Wo^T + bo.  O_view == Obf buffer row-major
// (the [b,h,d,n] element order IS the reference raw view to [b,n,c]).
// A tile staged via async global->LDS (ASYNCcnt) when the builtin exists.
__global__ __launch_bounds__(256) void out_kernel(
    const __bf16* __restrict__ Obf, const __bf16* __restrict__ Wob,
    const float* __restrict__ bo, float* __restrict__ Y) {
  __shared__ __bf16 As[2][64][40];
  const int lane = threadIdx.x & 31;
  const int wave = threadIdx.x >> 5;
  const int row0 = blockIdx.x * 64;
  const int col0 = blockIdx.y * 256 + wave * 32;

  const int r = threadIdx.x >> 2;
  const int c = (threadIdx.x & 3) * 8;
  const __bf16* arow = Obf + (size_t)(row0 + r) * 1024 + c;

  v8f acc[4][2] = {};

  // stage K-tile 0 into buffer 0 (16B per thread)
#ifdef HAVE_ASYNC_LDS
  __builtin_amdgcn_global_load_async_to_lds_b128(
      (v4i_vs*)arow, (v4i_vs*)&As[0][r][c], 0, 0);
  asm volatile("s_wait_asynccnt 0x0" ::: "memory");
#else
  *(v8bf*)&As[0][r][c] = *(const v8bf*)arow;
#endif
  __syncthreads();

  for (int k0 = 0; k0 < 1024; k0 += 32) {
    const int buf = (k0 >> 5) & 1;
    const bool more = (k0 + 32) < 1024;
    if (more) {  // stage next tile into other buffer, overlapped with wmma
#ifdef HAVE_ASYNC_LDS
      __builtin_amdgcn_global_load_async_to_lds_b128(
          (v4i_vs*)(arow + k0 + 32), (v4i_vs*)&As[buf ^ 1][r][c], 0, 0);
#else
      *(v8bf*)&As[buf ^ 1][r][c] = *(const v8bf*)(arow + k0 + 32);
#endif
    }
    v16bf b0 = load_frag_bf16(Wob, 1024, col0, k0);
    v16bf b1 = load_frag_bf16(Wob, 1024, col0 + 16, k0);
#pragma unroll
    for (int mi = 0; mi < 4; ++mi) {
      v16bf a = load_frag_bf16(&As[buf][0][0], 40, mi * 16, 0);
      acc[mi][0] = wmma_bf16(a, b0, acc[mi][0]);
      acc[mi][1] = wmma_bf16(a, b1, acc[mi][1]);
    }
#ifdef HAVE_ASYNC_LDS
    if (more) asm volatile("s_wait_asynccnt 0x0" ::: "memory");
#endif
    __syncthreads();
  }

  const int rr0 = (lane >> 4) << 3;
#pragma unroll
  for (int nj = 0; nj < 2; ++nj) {
    const int col = col0 + nj * 16 + (lane & 15);
    const float bias = bo[col];
#pragma unroll
    for (int mi = 0; mi < 4; ++mi)
#pragma unroll
      for (int i = 0; i < 8; ++i)
        Y[(size_t)(row0 + mi * 16 + rr0 + i) * 1024 + col] = acc[mi][nj][i] + bias;
  }
}

// ---------------------------------------------------------------------------
extern "C" void kernel_launch(void* const* d_in, const int* in_sizes, int n_in,
                              void* d_out, int out_size, void* d_ws, size_t ws_size,
                              hipStream_t stream) {
  const float* x     = (const float*)d_in[0];
  const float* Wq    = (const float*)d_in[1];
  const float* Wk    = (const float*)d_in[2];
  const float* Wv    = (const float*)d_in[3];
  const float* scale = (const float*)d_in[4];
  const float* Wo    = (const float*)d_in[5];
  const float* bo    = (const float*)d_in[6];
  float* Y = (float*)d_out;

  char* ws = (char*)d_ws;
  const size_t MB = 1024 * 1024;
  __bf16* Qt   = (__bf16*)(ws + 0);          //  64 MB [b,h,d,n]
  __bf16* Kt   = (__bf16*)(ws + 64 * MB);    //  64 MB [b,h,d,n]
  __bf16* Vt   = (__bf16*)(ws + 128 * MB);   //  64 MB [b,h,n,d]
  __bf16* Obf  = (__bf16*)(ws + 192 * MB);   //  64 MB [b,h,d,n]
  __bf16* Wqkv = (__bf16*)(ws + 256 * MB);   //   6 MB [3072 x 1024]
  __bf16* Wob  = (__bf16*)(ws + 262 * MB);   //   2 MB [1024 x 1024]
  float*  invq = (float*)(ws + 264 * MB);    //  32 KB
  float*  invk = invq + 8192;                //  32 KB

  cvt_w_kernel<<<dim3(1024, 4), 256, 0, stream>>>(Wq, Wk, Wv, Wo, Wqkv, Wob);
  proj_kernel<<<dim3(512, 12), 256, 0, stream>>>(x, Wqkv, Qt, Kt, Vt);
  norm_kernel<<<dim3(2048), 256, 0, stream>>>(Qt, Kt, invq, invk);
  attn_kernel<<<dim3(128), 256, 0, stream>>>(Qt, Kt, Vt, invq, invk, scale, Obf);
  out_kernel<<<dim3(512, 4), 256, 0, stream>>>(Obf, Wob, bo, Y);
}
